// VSN_82197084111231
// MI455X (gfx1250) — matrix operand order repaired and checked by reading, not compiled, not run on hardware
//
#include <hip/hip_runtime.h>
#include <hip/hip_bf16.h>

typedef __attribute__((ext_vector_type(16))) __bf16 v16bf;
typedef __attribute__((ext_vector_type(8)))  float  v8f;

#define BT   16384   // B*T
#define DD   256
#define NVV  16
#define DIN  4096

// Use CDNA5 async global->LDS copies (ASYNCcnt) for tile staging.
#define ASYNC_STAGE 1

union FragAB { v16bf v; uint4 q[2]; };

// 16-byte global -> LDS copy. Async path: GLOBAL_LOAD_ASYNC_TO_LDS_B128,
// per-lane LDS byte address in a VGPR (low 32 bits of the flat LDS pointer
// are the LDS offset), 64-bit global address in a VGPR pair.
__device__ __forceinline__ void stage16(const __bf16* g, __bf16* l) {
#if ASYNC_STAGE
    unsigned laddr = (unsigned)(size_t)(void*)l;
    asm volatile("global_load_async_to_lds_b128 %0, %1, off"
                 :: "v"(laddr), "v"(g) : "memory");
#else
    *(uint4*)l = *(const uint4*)g;
#endif
}

__device__ __forceinline__ void stage_wait() {
#if ASYNC_STAGE
    asm volatile("s_wait_asynccnt 0x0" ::: "memory");
#endif
}

// ---------------------------------------------------------------------------
// f32 -> bf16 (RNE), contiguous
// ---------------------------------------------------------------------------
__global__ void f32_to_bf16_kernel(const float* __restrict__ in,
                                   __bf16* __restrict__ out, long n) {
    long i = ((long)blockIdx.x * blockDim.x + threadIdx.x) * 4;
    if (i + 3 < n) {
        float4 f = *(const float4*)(in + i);
        out[i + 0] = (__bf16)f.x;
        out[i + 1] = (__bf16)f.y;
        out[i + 2] = (__bf16)f.z;
        out[i + 3] = (__bf16)f.w;
    } else {
        for (; i < n; ++i) out[i] = (__bf16)in[i];
    }
}

// f32 [K x N] row-major -> bf16 [N x K] (transposed), batched over z
__global__ void cvtT_kernel(const float* __restrict__ in,
                            __bf16* __restrict__ out, int K, int N) {
    size_t z = blockIdx.z;
    in  += z * (size_t)K * N;
    out += z * (size_t)K * N;
    int k = blockIdx.x * 16 + threadIdx.x;
    int n = blockIdx.y * 16 + threadIdx.y;
    out[(size_t)n * K + k] = (__bf16)in[(size_t)k * N + n];
}

// ---------------------------------------------------------------------------
// Tiled bf16 GEMM, f32 accum via v_wmma_f32_16x16x32_bf16.
// Block = 128 threads (4 waves). Tile: 64(M) x 64(N), BK = 32.
// B is given PRE-TRANSPOSED: Bt[N x K] row-major -> contiguous staging.
// ACT: 0 = none, 1 = ELU.  C = act(A@B + bias) stored bf16.
// ---------------------------------------------------------------------------
template<int ACT>
__global__ void gemm_bf16_kernel(const __bf16* __restrict__ A, long lda, long aOffZ,
                                 const __bf16* __restrict__ Bt, long ldbt, long bOffZ,
                                 const float*  __restrict__ bias, long biasOffZ,
                                 __bf16* __restrict__ C, long ldc, long cOffZ,
                                 int K) {
    const int tid  = threadIdx.x;
    const int wave = tid >> 5, lane = tid & 31;
    const int half = lane >> 4, mr = lane & 15;
    const long z   = blockIdx.z;
    A    += z * aOffZ;
    Bt   += z * bOffZ;
    bias += z * biasOffZ;
    C    += z * cOffZ;
    const int rowbase = blockIdx.y * 64;
    const int colbase = blockIdx.x * 64;

    __shared__ __align__(16) __bf16 As[64 * 40];   // padded stride 40
    __shared__ __align__(16) __bf16 Bs[64 * 40];   // [n][k] slice of Bt

    v8f acc[4] = {};

    for (int k0 = 0; k0 < K; k0 += 32) {
        __syncthreads();
        #pragma unroll
        for (int i = 0; i < 2; ++i) {
            int cid = tid * 2 + i;              // 0..255
            int r = cid >> 2, kc = (cid & 3) * 8;
            stage16(&A [(size_t)(rowbase + r) * lda  + k0 + kc], &As[r * 40 + kc]);
            stage16(&Bt[(size_t)(colbase + r) * ldbt + k0 + kc], &Bs[r * 40 + kc]);
        }
        if (k0 + 32 < K)
            __builtin_prefetch(&A[(size_t)(rowbase + (tid >> 1)) * lda + k0 + 32], 0, 0);
        stage_wait();
        __syncthreads();

        FragAB fa, fb;
        // A frag (16x32): lane holds row mr, contiguous K-run [16h,+16)
        fa.q[0] = *(const uint4*)&As[(wave * 16 + mr) * 40 + half * 16];
        fa.q[1] = *(const uint4*)&As[(wave * 16 + mr) * 40 + half * 16 + 8];
        #pragma unroll
        for (int c = 0; c < 4; ++c) {
            // B frag (32x16): lane holds col mr, contiguous K-run [16h,+16)
            fb.q[0] = *(const uint4*)&Bs[(c * 16 + mr) * 40 + half * 16];
            fb.q[1] = *(const uint4*)&Bs[(c * 16 + mr) * 40 + half * 16 + 8];
            acc[c] = __builtin_amdgcn_wmma_f32_16x16x32_bf16(
                false, fa.v, false, fb.v, (short)0, acc[c], false, false);
        }
    }

    #pragma unroll
    for (int c = 0; c < 4; ++c) {
        int col = colbase + c * 16 + mr;
        float bia = bias[col];
        #pragma unroll
        for (int r = 0; r < 8; ++r) {
            int row = rowbase + wave * 16 + half * 8 + r;
            float v = acc[c][r] + bia;
            if (ACT == 1) v = v > 0.f ? v : (__expf(v) - 1.f);
            C[(size_t)row * ldc + col] = (__bf16)v;
        }
    }
}

// ---------------------------------------------------------------------------
// GLU GEMM: a = A@B[:,n]+bias[n], b = A@B[:,n+256]+bias[n+256]
// out = bf16( a*sigmoid(b) + skip ).  Bt is [512 x K] transposed.
// ---------------------------------------------------------------------------
__global__ void gemm_glu_skip_kernel(const __bf16* __restrict__ A, long lda, long aOffZ,
                                     const __bf16* __restrict__ Bt, long ldbt, long bOffZ,
                                     const float*  __restrict__ bias, long biasOffZ,
                                     const float*  __restrict__ skip, long skipLd, long skipOffZ,
                                     __bf16* __restrict__ C, long ldc, long cOffZ,
                                     int K) {
    const int tid  = threadIdx.x;
    const int wave = tid >> 5, lane = tid & 31;
    const int half = lane >> 4, mr = lane & 15;
    const long z   = blockIdx.z;
    A    += z * aOffZ;
    Bt   += z * bOffZ;
    bias += z * biasOffZ;
    skip += z * skipOffZ;
    C    += z * cOffZ;
    const int rowbase = blockIdx.y * 64;
    const int colbase = blockIdx.x * 64;          // 0..255 ("a" half)

    __shared__ __align__(16) __bf16 As [64 * 40];
    __shared__ __align__(16) __bf16 Bsa[64 * 40];
    __shared__ __align__(16) __bf16 Bsb[64 * 40];

    v8f acca[4] = {};
    v8f accb[4] = {};

    for (int k0 = 0; k0 < K; k0 += 32) {
        __syncthreads();
        #pragma unroll
        for (int i = 0; i < 2; ++i) {
            int cid = tid * 2 + i;
            int r = cid >> 2, kc = (cid & 3) * 8;
            stage16(&A [(size_t)(rowbase + r)       * lda  + k0 + kc], &As [r * 40 + kc]);
            stage16(&Bt[(size_t)(colbase + r)       * ldbt + k0 + kc], &Bsa[r * 40 + kc]);
            stage16(&Bt[(size_t)(colbase + 256 + r) * ldbt + k0 + kc], &Bsb[r * 40 + kc]);
        }
        stage_wait();
        __syncthreads();

        FragAB fa, fb;
        fa.q[0] = *(const uint4*)&As[(wave * 16 + mr) * 40 + half * 16];
        fa.q[1] = *(const uint4*)&As[(wave * 16 + mr) * 40 + half * 16 + 8];
        #pragma unroll
        for (int c = 0; c < 4; ++c) {
            fb.q[0] = *(const uint4*)&Bsa[(c * 16 + mr) * 40 + half * 16];
            fb.q[1] = *(const uint4*)&Bsa[(c * 16 + mr) * 40 + half * 16 + 8];
            acca[c] = __builtin_amdgcn_wmma_f32_16x16x32_bf16(
                false, fa.v, false, fb.v, (short)0, acca[c], false, false);
            fb.q[0] = *(const uint4*)&Bsb[(c * 16 + mr) * 40 + half * 16];
            fb.q[1] = *(const uint4*)&Bsb[(c * 16 + mr) * 40 + half * 16 + 8];
            accb[c] = __builtin_amdgcn_wmma_f32_16x16x32_bf16(
                false, fa.v, false, fb.v, (short)0, accb[c], false, false);
        }
    }

    #pragma unroll
    for (int c = 0; c < 4; ++c) {
        int col = colbase + c * 16 + mr;
        float ba = bias[col], bb = bias[col + 256];
        #pragma unroll
        for (int r = 0; r < 8; ++r) {
            int row = rowbase + wave * 16 + half * 8 + r;
            float av = acca[c][r] + ba;
            float bv = accb[c][r] + bb;
            float g  = av * (1.f / (1.f + __expf(-bv)));
            float sk = skip[(size_t)row * skipLd + col];
            C[(size_t)row * ldc + col] = (__bf16)(g + sk);
        }
    }
}

// ---------------------------------------------------------------------------
// skip = flat @ Wgs + bgs  (K=4096, N=16), one block (256 thr) per token row
// ---------------------------------------------------------------------------
__global__ void skip_kernel(const float* __restrict__ vars,
                            const float* __restrict__ Wgs,
                            const float* __restrict__ bgs,
                            float* __restrict__ skipb) {
    int row = blockIdx.x, t = threadIdx.x;
    float p[16];
    #pragma unroll
    for (int c = 0; c < 16; ++c) p[c] = 0.f;
    for (int kb = 0; kb < 16; ++kb) {
        int k = kb * 256 + t;
        float x = vars[(size_t)row * DIN + k];
        const float* wr = &Wgs[(size_t)k * 16];
        #pragma unroll
        for (int c = 0; c < 16; ++c) p[c] += x * wr[c];
    }
    __shared__ float red[16][256];
    #pragma unroll
    for (int c = 0; c < 16; ++c) red[c][t] = p[c];
    __syncthreads();
    for (int s = 128; s; s >>= 1) {
        if (t < s) {
            #pragma unroll
            for (int c = 0; c < 16; ++c) red[c][t] += red[c][t + s];
        }
        __syncthreads();
    }
    if (t < 16) skipb[(size_t)row * 16 + t] = red[t][0] + bgs[t];
}

// ---------------------------------------------------------------------------
// weight head: h2 @ Wg3 (+bg3) -> GLU -> +skip -> LN(16) -> softmax(16)
// one wave per token; lane = output column (0..31). W3t is [32 x 256].
// ---------------------------------------------------------------------------
__global__ void wlogit_kernel(const __bf16* __restrict__ h2,
                              const __bf16* __restrict__ W3t,
                              const float* __restrict__ b3,
                              const float* __restrict__ skipb,
                              const float* __restrict__ gg,
                              const float* __restrict__ bgn,
                              float* __restrict__ wout) {
    int row = blockIdx.x, lane = threadIdx.x;
    float acc = 0.f;
    for (int k = 0; k < DD; ++k)
        acc += (float)h2[(size_t)row * DD + k] * (float)W3t[(size_t)lane * DD + k];
    acc += b3[lane];
    float bpart = __shfl(acc, lane + 16, 32);           // lanes<16 get "b" half
    float x = acc * (1.f / (1.f + __expf(-bpart)));     // GLU
    x += skipb[(size_t)row * 16 + (lane & 15)];
    float s = x, s2 = x * x;
    for (int o = 8; o; o >>= 1) { s += __shfl_xor(s, o, 16); s2 += __shfl_xor(s2, o, 16); }
    float m   = s * (1.f / 16.f);
    float var = s2 * (1.f / 16.f) - m * m;
    float xn  = (x - m) * rsqrtf(var + 1e-6f) * gg[lane & 15] + bgn[lane & 15];
    float mx = xn;
    for (int o = 8; o; o >>= 1) mx = fmaxf(mx, __shfl_xor(mx, o, 16));
    float e = __expf(xn - mx);
    float se = e;
    for (int o = 8; o; o >>= 1) se += __shfl_xor(se, o, 16);
    if (lane < 16) wout[(size_t)row * 16 + lane] = e / se;
}

// ---------------------------------------------------------------------------
// final: per token, per variable LayerNorm(256) then softmax-weighted combine
// ---------------------------------------------------------------------------
__global__ void combine_kernel(const __bf16* __restrict__ gskip,
                               const float* __restrict__ wbuf,
                               const float* __restrict__ gv,
                               const float* __restrict__ bvn,
                               float* __restrict__ out) {
    int row = blockIdx.x, d = threadIdx.x;   // 256 threads
    __shared__ float ssum[8], ssq[8], sstat[2];
    float acc = 0.f;
    for (int v = 0; v < NVV; ++v) {
        float x = (float)gskip[((size_t)v * BT + row) * DD + d];
        float s = x, s2 = x * x;
        for (int o = 16; o; o >>= 1) { s += __shfl_xor(s, o, 32); s2 += __shfl_xor(s2, o, 32); }
        if ((d & 31) == 0) { ssum[d >> 5] = s; ssq[d >> 5] = s2; }
        __syncthreads();
        if (d == 0) {
            float ts = 0.f, tq = 0.f;
            for (int i = 0; i < 8; ++i) { ts += ssum[i]; tq += ssq[i]; }
            float m = ts / 256.f;
            sstat[0] = m;
            sstat[1] = rsqrtf(tq / 256.f - m * m + 1e-6f);
        }
        __syncthreads();
        float xn = (x - sstat[0]) * sstat[1] * gv[v * DD + d] + bvn[v * DD + d];
        acc += wbuf[(size_t)row * 16 + v] * xn;
        __syncthreads();
    }
    out[(size_t)row * DD + d] = acc;
}

// ---------------------------------------------------------------------------
extern "C" void kernel_launch(void* const* d_in, const int* in_sizes, int n_in,
                              void* d_out, int out_size, void* d_ws, size_t ws_size,
                              hipStream_t stream) {
    (void)in_sizes; (void)n_in; (void)out_size; (void)ws_size;

    const float* vars = (const float*)d_in[0];
    const float* Wg1  = (const float*)d_in[1];
    const float* bg1  = (const float*)d_in[2];
    const float* Wg2  = (const float*)d_in[3];
    const float* bg2  = (const float*)d_in[4];
    const float* Wg3  = (const float*)d_in[5];
    const float* bg3  = (const float*)d_in[6];
    const float* Wgs  = (const float*)d_in[7];
    const float* bgs  = (const float*)d_in[8];
    const float* gg   = (const float*)d_in[9];
    const float* bgn  = (const float*)d_in[10];
    const float* Wv1  = (const float*)d_in[11];
    const float* bv1  = (const float*)d_in[12];
    const float* Wv2  = (const float*)d_in[13];
    const float* bv2  = (const float*)d_in[14];
    const float* Wv3  = (const float*)d_in[15];
    const float* bv3  = (const float*)d_in[16];
    const float* gv   = (const float*)d_in[17];
    const float* bvn  = (const float*)d_in[18];

    float* out  = (float*)d_out;
    float* wout = out + (size_t)BT * DD;     // weight output segment

    // workspace layout (bf16 element offsets); all weight buffers TRANSPOSED
    __bf16* wsb  = (__bf16*)d_ws;
    __bf16* Wg1T = wsb;                          // [256 x 4096]
    __bf16* Wg2T = Wg1T + 1048576;               // [256 x 256]
    __bf16* Wg3T = Wg2T + 65536;                 // [32 x 256]
    __bf16* Wv1T = Wg3T + 8192;                  // 16 x [256 x 256]
    __bf16* Wv2T = Wv1T + 1048576;               // 16 x [256 x 256]
    __bf16* Wv3T = Wv2T + 1048576;               // 16 x [512 x 256]
    __bf16* h1   = Wv3T + 2097152;               // 4194304
    __bf16* h2   = h1 + 4194304;                 // 4194304
    __bf16* Vb   = h2 + 4194304;                 // 67108864 (reused as hv2)
    __bf16* hv1  = Vb + 67108864;                // 67108864 (reused as gskip)
    float* skipb = (float*)(hv1 + 67108864);     // 262144 f32

    __bf16* hv2   = Vb;    // alias: Vb dead after per-var GEMM1
    __bf16* gskip = hv1;   // alias: hv1 dead after per-var GEMM2

    dim3 t16(16, 16);
    cvtT_kernel<<<dim3(DIN / 16, DD / 16, 1),  t16, 0, stream>>>(Wg1, Wg1T, DIN, DD);
    cvtT_kernel<<<dim3(DD / 16,  DD / 16, 1),  t16, 0, stream>>>(Wg2, Wg2T, DD, DD);
    cvtT_kernel<<<dim3(DD / 16,  32 / 16, 1),  t16, 0, stream>>>(Wg3, Wg3T, DD, 32);
    cvtT_kernel<<<dim3(DD / 16,  DD / 16, NVV), t16, 0, stream>>>(Wv1, Wv1T, DD, DD);
    cvtT_kernel<<<dim3(DD / 16,  DD / 16, NVV), t16, 0, stream>>>(Wv2, Wv2T, DD, DD);
    cvtT_kernel<<<dim3(DD / 16, 512 / 16, NVV), t16, 0, stream>>>(Wv3, Wv3T, DD, 512);

    long nvb = (long)BT * DIN;
    f32_to_bf16_kernel<<<(int)((nvb / 4 + 255) / 256), 256, 0, stream>>>(vars, Vb, nvb);

    // weight-network skip: flat @ Wgs + bgs
    skip_kernel<<<BT, 256, 0, stream>>>(vars, Wgs, bgs, skipb);

    dim3 blk(128);
    // weight GRN: h1 = ELU(flat @ Wg1 + bg1)   [K=4096]
    gemm_bf16_kernel<1><<<dim3(DD / 64, BT / 64, 1), blk, 0, stream>>>(
        Vb, DIN, 0, Wg1T, DIN, 0, bg1, 0, h1, DD, 0, DIN);
    // h2 = h1 @ Wg2 + bg2
    gemm_bf16_kernel<0><<<dim3(DD / 64, BT / 64, 1), blk, 0, stream>>>(
        h1, DD, 0, Wg2T, DD, 0, bg2, 0, h2, DD, 0, DD);
    // wlogits -> softmax weights
    wlogit_kernel<<<BT, 32, 0, stream>>>(h2, Wg3T, bg3, skipb, gg, bgn, wout);

    // per-variable GRNs (grid.z = variable)
    gemm_bf16_kernel<1><<<dim3(DD / 64, BT / 64, NVV), blk, 0, stream>>>(
        Vb, DIN, 256, Wv1T, DD, 65536, bv1, 256, hv1, DD, (long)BT * DD, DD);
    gemm_bf16_kernel<0><<<dim3(DD / 64, BT / 64, NVV), blk, 0, stream>>>(
        hv1, DD, (long)BT * DD, Wv2T, DD, 65536, bv2, 256, hv2, DD, (long)BT * DD, DD);
    gemm_glu_skip_kernel<<<dim3(DD / 64, BT / 64, NVV), blk, 0, stream>>>(
        hv2, DD, (long)BT * DD, Wv3T, DD, 131072, bv3, 512,
        vars, DIN, 256, gskip, DD, (long)BT * DD, DD);

    // LN(256) per variable + softmax-weighted combine
    combine_kernel<<<BT, 256, 0, stream>>>(gskip, wout, gv, bvn, out);
}
